// OmniMVS_37005438222787
// MI455X (gfx1250) — matrix operand order repaired and checked by reading, not compiled, not run on hardware
//
#include <hip/hip_runtime.h>
#include <stdint.h>

typedef __bf16 bf16;
typedef __attribute__((ext_vector_type(16))) __bf16 v16bf;
typedef __attribute__((ext_vector_type(8)))  float  v8f;

union FragBF { v16bf v; uint4 q[2]; };

// ---------------------------------------------------------------- zero fill
__global__ void zero_u4(uint4* __restrict__ p, size_t n) {
    size_t i = (size_t)blockIdx.x * blockDim.x + threadIdx.x;
    if (i < n) { uint4 z; z.x = z.y = z.z = z.w = 0u; p[i] = z; }
}

// ------------------------------------------------- weight transform to bf16
// w: [Cout][Cin][3][3][3] f32  ->  A: [Cout][tap*Cin + ci] bf16
__global__ void prep_w_kernel(const float* __restrict__ w, bf16* __restrict__ A,
                              int Cout, int Cin) {
    int idx = blockIdx.x * blockDim.x + threadIdx.x;
    int total = Cout * Cin * 27;
    if (idx >= total) return;
    int co  = idx / (Cin * 27);
    int rem = idx % (Cin * 27);
    int ci  = rem / 27;
    int t   = rem % 27;
    A[(size_t)co * (Cin * 27) + (size_t)t * Cin + ci] = (bf16)w[idx];
}

// --------------------------------------------- 2D feature conv (3->32, s2) + ReLU
// in: [3][640][1280] f32, out feat[cam]: [320][640][32] f32 channels-last
__global__ void feat_conv2d(const float* __restrict__ cam1, const float* __restrict__ cam2,
                            const float* __restrict__ cam3, const float* __restrict__ cam4,
                            const float* __restrict__ wfeat, float* __restrict__ feat) {
    __shared__ float wl[864];                       // [32][3][3][3]
    for (int i = threadIdx.x; i < 864; i += 256) wl[i] = wfeat[i];
    __syncthreads();
    const int cam = blockIdx.z;
    const float* in = (cam == 0) ? cam1 : (cam == 1) ? cam2 : (cam == 2) ? cam3 : cam4;
    int pix = blockIdx.x * 256 + threadIdx.x;       // 320*640 = 204800
    int oy = pix / 640, ox = pix % 640;
    float acc[32];
#pragma unroll
    for (int i = 0; i < 32; ++i) acc[i] = 0.f;
    for (int c = 0; c < 3; ++c)
        for (int ky = 0; ky < 3; ++ky) {
            int iy = oy * 2 + ky - 1;
            if (iy < 0 || iy >= 640) continue;
            for (int kx = 0; kx < 3; ++kx) {
                int ix = ox * 2 + kx - 1;
                if (ix < 0 || ix >= 1280) continue;
                float v = in[(size_t)c * 640 * 1280 + (size_t)iy * 1280 + ix];
                const float* wp = &wl[(c * 3 + ky) * 3 + kx];
#pragma unroll
                for (int co = 0; co < 32; ++co) acc[co] = fmaf(v, wp[co * 27], acc[co]);
            }
        }
    float* op = feat + ((size_t)(cam * 320 + oy) * 640 + ox) * 32;
#pragma unroll
    for (int co = 0; co < 32; ++co) op[co] = fmaxf(acc[co], 0.f);
}

// ------------------- fused grid_sample + bilinear resize -> padded bf16 cost volume
// feat: [4][320][640][32] f32, grids: [4][4][320][640][2] f32
// costs: [6][258][258][128] bf16 (zero halo), channel = cam*32 + c
__global__ void warp_resize(const float* __restrict__ feat, const float* __restrict__ grids,
                            bf16* __restrict__ costs) {
    int pix = blockIdx.x * 256 + threadIdx.x;       // 256*256
    int oy = pix >> 8, ox = pix & 255;
    int d = blockIdx.y, cam = blockIdx.z;
    float sy = (oy + 0.5f) * (320.f / 256.f) - 0.5f;
    float sx = (ox + 0.5f) * (640.f / 256.f) - 0.5f;
    float fy0 = floorf(sy), fx0 = floorf(sx);
    int y0 = (int)fy0, x0 = (int)fx0;
    float fy = sy - fy0, fx = sx - fx0;
    float acc[32];
#pragma unroll
    for (int i = 0; i < 32; ++i) acc[i] = 0.f;
    const float* fbase = feat + (size_t)cam * 320 * 640 * 32;
#pragma unroll
    for (int ny = 0; ny < 2; ++ny)
#pragma unroll
    for (int nx = 0; nx < 2; ++nx) {
        int wy = min(max(y0 + ny, 0), 319);
        int wx = min(max(x0 + nx, 0), 639);
        float wr = (ny ? fy : 1.f - fy) * (nx ? fx : 1.f - fx);
        const float* g = grids + ((((size_t)cam * 4 + d) * 320 + wy) * 640 + wx) * 2;
        float fpx = (g[0] + 1.f) * 320.f - 0.5f;    // W=640
        float fpy = (g[1] + 1.f) * 160.f - 0.5f;    // H=320
        float gx0 = floorf(fpx), gy0 = floorf(fpy);
        int px0 = (int)gx0, py0 = (int)gy0;
        float ax = fpx - gx0, ay = fpy - gy0;
#pragma unroll
        for (int ty = 0; ty < 2; ++ty)
#pragma unroll
        for (int tx = 0; tx < 2; ++tx) {
            int sxp = px0 + tx, syp = py0 + ty;
            bool valid = (sxp >= 0) && (sxp < 640) && (syp >= 0) && (syp < 320);
            float wgt = wr * (ty ? ay : 1.f - ay) * (tx ? ax : 1.f - ax);
            if (valid) {
                const float4* fp = (const float4*)(fbase + ((size_t)syp * 640 + sxp) * 32);
#pragma unroll
                for (int i = 0; i < 8; ++i) {
                    float4 v = fp[i];
                    acc[4 * i + 0] = fmaf(wgt, v.x, acc[4 * i + 0]);
                    acc[4 * i + 1] = fmaf(wgt, v.y, acc[4 * i + 1]);
                    acc[4 * i + 2] = fmaf(wgt, v.z, acc[4 * i + 2]);
                    acc[4 * i + 3] = fmaf(wgt, v.w, acc[4 * i + 3]);
                }
            }
        }
    }
    bf16* op = costs + (((size_t)(d + 1) * 258 + (oy + 1)) * 258 + (ox + 1)) * 128 + cam * 32;
#pragma unroll
    for (int i = 0; i < 32; ++i) op[i] = (bf16)acc[i];
}

// ---------------------------------- implicit-GEMM 3D conv + ReLU via bf16 WMMA
// X: [6][258][258][CIN] bf16 padded, Wg: [COUT][27*CIN] bf16, Y: [6][258][258][COUT]
// Tile: 64 output x-positions per block, full Cout. 256 thr = 8 waves.
// B-tiles staged into double-buffered LDS with GLOBAL_LOAD_ASYNC_TO_LDS_B128,
// overlapping the tap t+1 DMA with tap t WMMA compute (sync: s_wait_asynccnt + barrier).
template <int CIN, int COUT>
__global__ void __launch_bounds__(256, 2)
conv3d_wmma(const bf16* __restrict__ X, const bf16* __restrict__ Wg, bf16* __restrict__ Y) {
    constexpr int KTOT = 27 * CIN;
    constexpr int MT   = COUT / 16;    // M tiles
    constexpr int WPM  = 8 / MT;       // waves per M tile
    constexpr int NTW  = 4 / WPM;      // N subtiles (of 16) per wave
    __shared__ __align__(16) bf16 smem[2][64 * CIN];

    const int tid  = threadIdx.x;
    const int lane = tid & 31;
    const int wave = tid >> 5;
    const int mtile   = wave / WPM;
    const int ntStart = (wave % WPM) * NTW;

    const int x0 = blockIdx.x * 64;
    const int y  = blockIdx.y;
    const int z  = blockIdx.z;

    v8f acc[NTW];
#pragma unroll
    for (int j = 0; j < NTW; ++j)
#pragma unroll
        for (int r = 0; r < 8; ++r) acc[j][r] = 0.f;

    // A-fragment addressing per documented 16-bit 16x32 layout
    const int row  = mtile * 16 + (lane & 15);
    const int kloA = (lane & 16) ? 8 : 0;   // lanes 16-31: elems 0-7 = K 8-15, elems 8-15 = K 24-31
    const int kloB = (lane & 16) ? 8 : 0;   // B: same K order as A (lanes 16-31: K 8-15, 24-31)
    const size_t rowBase = (size_t)row * KTOT;

    // Issue one tap's 64*CIN bf16 block as async memory->LDS b128 copies.
    auto issue_tap = [&](int t, int buf) {
        const int dz = t / 9 - 1, dy = (t / 3) % 3 - 1, dx = t % 3 - 1;
        const bf16* src = X + (((size_t)(z + 1 + dz) * 258 + (y + 1 + dy)) * 258 +
                               (x0 + 1 + dx)) * (size_t)CIN;
        const uint32_t ldsBase = (uint32_t)(uintptr_t)(&smem[buf][0]);
        constexpr int CH16 = 64 * CIN / 8;             // 16B chunks
#pragma unroll
        for (int i = 0; i < CH16 / 256; ++i) {
            const int c = tid + i * 256;
            const uint32_t ldsAddr = ldsBase + (uint32_t)c * 16u;
            const bf16* g = src + c * 8;
            asm volatile("global_load_async_to_lds_b128 %0, %1, off"
                         :: "v"(ldsAddr), "v"(g) : "memory");
        }
    };

    issue_tap(0, 0);

    for (int t = 0; t < 27; ++t) {
        asm volatile("s_wait_asynccnt 0x0" ::: "memory");   // own async portion landed
        __syncthreads();                                    // everyone's portion landed
        if (t + 1 < 27) issue_tap(t + 1, (t + 1) & 1);      // overlap next tap's DMA
        const bf16* sb = &smem[t & 1][0];
#pragma unroll
        for (int kk = 0; kk < CIN / 32; ++kk) {
            const int kbase = t * CIN + kk * 32;
            FragBF a;
            const bf16* ap = Wg + rowBase + kbase + kloA;
            a.q[0] = *(const uint4*)(ap);
            a.q[1] = *(const uint4*)(ap + 16);
#pragma unroll
            for (int j = 0; j < NTW; ++j) {
                const int col = (ntStart + j) * 16 + (lane & 15);
                FragBF b;
                const bf16* bp = sb + (size_t)col * CIN + kk * 32 + kloB;
                b.q[0] = *(const uint4*)(bp);
                b.q[1] = *(const uint4*)(bp + 16);
                acc[j] = __builtin_amdgcn_wmma_f32_16x16x32_bf16(
                    false, a.v, false, b.v, (short)0, acc[j], false, false);
            }
        }
    }

    // Store: VGPR r -> channel mrow + r, col = lane&15 within subtile. ReLU + bf16.
    const int mrow = mtile * 16 + ((lane & 16) ? 8 : 0);
#pragma unroll
    for (int j = 0; j < NTW; ++j) {
        const int col = (ntStart + j) * 16 + (lane & 15);
        bf16* op = Y + (((size_t)(z + 1) * 258 + (y + 1)) * 258 + (x0 + col + 1)) * (size_t)COUT + mrow;
#pragma unroll
        for (int r = 0; r < 8; ++r) op[r] = (bf16)fmaxf(acc[j][r], 0.f);
    }
}

// --------------------------------- last conv 32->1 (no relu), direct VALU
// X: [6][258][258][32] bf16 padded, w: [32][27] f32 (OIDHW flattened), out3: [4][256][256] f32
__global__ void conv3d_last(const bf16* __restrict__ X, const float* __restrict__ w,
                            float* __restrict__ out3) {
    __shared__ float wl[864];
    for (int i = threadIdx.x; i < 864; i += 256) wl[i] = w[i];
    __syncthreads();
    int idx = blockIdx.x * 256 + threadIdx.x;      // 262144
    int x = idx & 255, y = (idx >> 8) & 255, z = idx >> 16;
    float acc = 0.f;
    for (int t = 0; t < 27; ++t) {
        int dz = t / 9 - 1, dy = (t / 3) % 3 - 1, dx = t % 3 - 1;
        const bf16* src = X + (((size_t)(z + 1 + dz) * 258 + (y + 1 + dy)) * 258 + (x + 1 + dx)) * 32;
#pragma unroll
        for (int ci = 0; ci < 32; ++ci) acc = fmaf((float)src[ci], wl[ci * 27 + t], acc);
    }
    out3[idx] = acc;
}

// ------------- depth resize 4->8 (bilinear) + softmax + disparity expectation
__global__ void final_softmax(const float* __restrict__ out3, float* __restrict__ out) {
    int idx = blockIdx.x * 256 + threadIdx.x;      // 65536
    float v[4];
#pragma unroll
    for (int d = 0; d < 4; ++d) v[d] = out3[(size_t)d * 65536 + idx];
    float e[8], mx = -1e30f;
#pragma unroll
    for (int od = 0; od < 8; ++od) {
        float sd = od * 0.5f - 0.25f;
        float fd = floorf(sd);
        int   d0 = (int)fd;
        float f  = sd - fd;
        int da = min(max(d0, 0), 3), db = min(max(d0 + 1, 0), 3);
        e[od] = v[da] * (1.f - f) + v[db] * f;
        mx = fmaxf(mx, e[od]);
    }
    float s = 0.f, num = 0.f;
#pragma unroll
    for (int od = 0; od < 8; ++od) {
        float p = __expf(e[od] - mx);
        s += p; num += p * (float)od;
    }
    out[idx] = num / s;
}

// ---------------------------------------------------------------- launcher
extern "C" void kernel_launch(void* const* d_in, const int* in_sizes, int n_in,
                              void* d_out, int out_size, void* d_ws, size_t ws_size,
                              hipStream_t stream) {
    (void)in_sizes; (void)n_in; (void)out_size; (void)ws_size;
    const float* cam1   = (const float*)d_in[0];
    const float* cam2   = (const float*)d_in[1];
    const float* cam3   = (const float*)d_in[2];
    const float* cam4   = (const float*)d_in[3];
    const float* grids  = (const float*)d_in[4];
    const float* w_feat = (const float*)d_in[5];
    const float* w_fus  = (const float*)d_in[6];
    const float* w_r1   = (const float*)d_in[7];
    const float* w_r2   = (const float*)d_in[8];

    char* ws = (char*)d_ws;
    size_t off = 0;
    auto take = [&](size_t bytes) -> char* {
        char* p = ws + off;
        off = (off + bytes + 255) & ~(size_t)255;
        return p;
    };
    float* feat  = (float*)take(4ull * 320 * 640 * 32 * 4);       // ~105 MB
    bf16*  costs = (bf16*) take(6ull * 258 * 258 * 128 * 2);      // ~102 MB padded
    bf16*  x1    = (bf16*) take(6ull * 258 * 258 * 64  * 2);      // ~51 MB padded
    bf16*  x2    = (bf16*) take(6ull * 258 * 258 * 32  * 2);      // ~26 MB padded
    float* out3  = (float*)take(4ull * 256 * 256 * 4);
    bf16*  w1    = (bf16*) take(64ull * 3456 * 2);
    bf16*  w2    = (bf16*) take(32ull * 1728 * 2);

    // Zero the three padded activation buffers (contiguous region) -> halos stay 0.
    size_t zbytes = (size_t)((char*)out3 - (char*)costs);
    size_t zn = zbytes / 16;
    zero_u4<<<(unsigned)((zn + 255) / 256), 256, 0, stream>>>((uint4*)costs, zn);

    prep_w_kernel<<<(64 * 128 * 27 + 255) / 256, 256, 0, stream>>>(w_fus, w1, 64, 128);
    prep_w_kernel<<<(32 * 64 * 27 + 255) / 256, 256, 0, stream>>>(w_r1, w2, 32, 64);

    feat_conv2d<<<dim3(800, 1, 4), 256, 0, stream>>>(cam1, cam2, cam3, cam4, w_feat, feat);
    warp_resize<<<dim3(256, 4, 4), 256, 0, stream>>>(feat, grids, costs);

    conv3d_wmma<128, 64><<<dim3(4, 256, 4), 256, 0, stream>>>(costs, w1, x1);
    conv3d_wmma<64, 32><<<dim3(4, 256, 4), 256, 0, stream>>>(x1, w2, x2);

    conv3d_last<<<1024, 256, 0, stream>>>(x2, w_r2, out3);
    final_softmax<<<256, 256, 0, stream>>>(out3, (float*)d_out);
}
